// MultiHeadAttention_41936060678620
// MI455X (gfx1250) — matrix operand order repaired and checked by reading, not compiled, not run on hardware
//
#include <hip/hip_runtime.h>
#include <hip/hip_bf16.h>
#include <stdint.h>

// ---------------------------------------------------------------------------
// Problem constants (from reference): B=8, S=2048, H=256, NH=4, D=64
// ---------------------------------------------------------------------------
constexpr int B_ = 8, S_ = 2048, H_ = 256, NH_ = 4, D_ = 64;
constexpr size_t MROWS = (size_t)B_ * S_;     // 16384
constexpr size_t XE    = MROWS * H_;          // 4,194,304 elements

typedef __attribute__((ext_vector_type(16))) __bf16 v16bf;
typedef __attribute__((ext_vector_type(2)))  __bf16 bf16x2;
typedef __attribute__((ext_vector_type(8)))  float  v8f;

static __device__ __forceinline__ v8f wmma_bf16(v16bf a, v16bf b, v8f c) {
  // D = A(16x32 bf16) x B(32x16 bf16) + C(16x16 f32)
  return __builtin_amdgcn_wmma_f32_16x16x32_bf16(
      /*neg_a=*/false, a, /*neg_b=*/false, b,
      /*c_mod=*/(short)0, c, /*reuse_a=*/false, /*reuse_b=*/false);
}

// A-fragment k index for element e (0..15), half-wave g (0/1):
//   lanes 0-15 hold K 0..7 (v0..3) and 16..23 (v4..7); lanes 16-31: 8..15 / 24..31
static __device__ __forceinline__ int a_kofs(int g, int e) {
  return g * 8 + (e < 8 ? e : e + 8);
}

// gfx1250 async copy: 16B global -> LDS, tracked by ASYNCcnt (inline asm is
// portable across ROCm-7.2 / amdgpu-toolchain builtin arity differences).
static __device__ __forceinline__ void async_ld16(uint32_t lds_off, const __bf16* g) {
  asm volatile("global_load_async_to_lds_b128 %0, %1, off"
               :: "v"(lds_off), "v"((uint64_t)(uintptr_t)g)
               : "memory");
}
static __device__ __forceinline__ void wait_async0() {
  asm volatile("s_wait_asynccnt 0x0" ::: "memory");
}

// ---------------------------------------------------------------------------
// Stage 0a: mixed = bf16(attr + x + pos); xb = bf16(x)
// ---------------------------------------------------------------------------
__global__ void __launch_bounds__(256)
prep_inputs_k(const float* __restrict__ x, const float* __restrict__ attr,
              const float* __restrict__ pos,
              __bf16* __restrict__ mixed, __bf16* __restrict__ xb, int n) {
  int i = blockIdx.x * 256 + threadIdx.x;
  if (i < n) {
    mixed[i] = (__bf16)(attr[i] + x[i] + pos[i]);
    xb[i]    = (__bf16)x[i];
  }
}

// Stage 0b: Wt[n][k] = bf16(W[k][n])  (so B-fragments read contiguous k pairs)
__global__ void __launch_bounds__(256)
prep_weight_k(const float* __restrict__ W, __bf16* __restrict__ Wt) {
  int i = blockIdx.x * 256 + threadIdx.x;       // 0..65535
  int n = i >> 8, k = i & 255;
  Wt[i] = (__bf16)(W[(size_t)k * H_ + n]);
}

// ---------------------------------------------------------------------------
// Stage 1/4: wave-tile WMMA GEMM.  Each wave owns a 32x16 output tile (two
// 16x16 C fragments sharing every weight B-fragment -> 2 wmma per B load).
//   A  : [M,K] bf16 row-major;  Wt : [N,K] bf16 (transposed weight)
// OUT_MODE 0: Q/K  -> bf16 [B,NH,S,D]
// OUT_MODE 1: V    -> bf16 [B,NH,D,S]  (transposed, contiguous along s)
// OUT_MODE 2: f32  [M,N] row-major (hidden, pre-layernorm)
// ---------------------------------------------------------------------------
template <int OUT_MODE>
__global__ void __launch_bounds__(256)
gemm_wmma_k(const __bf16* __restrict__ A, const __bf16* __restrict__ Wt,
            const float* __restrict__ bias, void* __restrict__ out) {
  constexpr int K = H_, N = H_;
  const int w    = (blockIdx.x * 256 + threadIdx.x) >> 5;   // global wave id
  const int lane = threadIdx.x & 31;
  const int g    = lane >> 4;
  const int ln   = lane & 15;
  const int tm   = (w >> 4) << 5;     // 32-row supertile; 16 n-tiles per row
  const int tn   = (w & 15) << 4;

  v8f acc0 = {}, acc1 = {};
  const __bf16* Arow0 = A  + (size_t)(tm + ln) * K;
  const __bf16* Arow1 = A  + (size_t)(tm + 16 + ln) * K;
  const __bf16* Wrow  = Wt + (size_t)(tn + ln) * K;
  for (int k0 = 0; k0 < K; k0 += 32) {
    v16bf af0, af1, bf;
#pragma unroll
    for (int v = 0; v < 8; ++v) {
      int ka = k0 + a_kofs(g, 2 * v);
      bf16x2 pa0 = *(const bf16x2*)(Arow0 + ka);
      af0[2 * v] = pa0[0]; af0[2 * v + 1] = pa0[1];
      bf16x2 pa1 = *(const bf16x2*)(Arow1 + ka);
      af1[2 * v] = pa1[0]; af1[2 * v + 1] = pa1[1];
      bf16x2 pb = *(const bf16x2*)(Wrow + ka);
      bf[2 * v] = pb[0]; bf[2 * v + 1] = pb[1];
    }
    acc0 = wmma_bf16(af0, bf, acc0);
    acc1 = wmma_bf16(af1, bf, acc1);
  }

  const int   n    = tn + ln;
  const float bval = bias[n];
#pragma unroll
  for (int half = 0; half < 2; ++half) {
    const v8f& acc = half ? acc1 : acc0;
    const int  tmh = tm + half * 16;
    if (OUT_MODE == 0) {                       // [B,NH,S,D]
      __bf16* o = (__bf16*)out;
      const int h = n >> 6, d = n & 63;
#pragma unroll
      for (int r = 0; r < 8; ++r) {
        int m = tmh + g * 8 + r, b = m >> 11, s = m & (S_ - 1);
        o[((((size_t)b * NH_ + h) * S_) + s) * D_ + d] = (__bf16)(acc[r] + bval);
      }
    } else if (OUT_MODE == 1) {                // [B,NH,D,S] (transposed V)
      __bf16* o = (__bf16*)out;
      const int h = n >> 6, d = n & 63;
      int m0 = tmh + g * 8, b = m0 >> 11, s0 = m0 & (S_ - 1);
      size_t base = (((size_t)b * NH_ + h) * D_ + d) * S_ + s0;
#pragma unroll
      for (int r = 0; r < 8; ++r) o[base + r] = (__bf16)(acc[r] + bval);
    } else {                                   // f32 [M,N]
      float* o = (float*)out;
#pragma unroll
      for (int r = 0; r < 8; ++r) {
        int m = tmh + g * 8 + r;
        o[(size_t)m * N + n] = acc[r] + bval;
      }
    }
  }
}

// ---------------------------------------------------------------------------
// Stage 2: flash attention with double-buffered async K/V staging.
// Block = 8 waves = 128 q rows of one (b,h). Key tiles of 64:
//   * 256 threads async-copy K tile (64x64 bf16) and V^T tile (64x64 bf16)
//     to LDS with global_load_async_to_lds_b128, double buffered so the
//     next tile's copy overlaps this tile's WMMA + softmax.
//   * per wave: 8 wmma (QK^T) -> f32 online softmax (16-lane shfl_xor row
//     reductions) -> P via per-wave LDS slab -> 8 wmma (P.V).
// LDS: 2*8KB (K) + 2*8KB (V) + 16KB (P) = 48KB.
// ---------------------------------------------------------------------------
__global__ void __launch_bounds__(256)
flash_attn_k(const __bf16* __restrict__ Q, const __bf16* __restrict__ K,
             const __bf16* __restrict__ Vt, const float* __restrict__ mask,
             __bf16* __restrict__ ctx) {
  __shared__ __bf16 k_tile[2][64][64];   // [buf][key][d]
  __shared__ __bf16 v_tile[2][64][64];   // [buf][d][key]
  __shared__ __bf16 p_tile[8][16][64];   // per-wave P tile

  const int tid  = threadIdx.x;
  const int wave = tid >> 5;
  const int lane = tid & 31;
  const int g    = lane >> 4;
  const int ln   = lane & 15;
  const int bh   = blockIdx.x >> 4;     // B*NH = 32
  const int qblk = blockIdx.x & 15;     // S/128 = 16
  const int b    = bh >> 2, h = bh & 3;
  const int tq   = qblk * 128 + wave * 16;

  const __bf16* Qp = Q  + ((size_t)bh * S_ + tq) * D_;
  const __bf16* Kp = K  + (size_t)bh * S_ * D_;
  const __bf16* Vp = Vt + (size_t)bh * D_ * S_;
  const float*  Mp = mask + ((size_t)b * S_ + tq) * S_;   // mask[b,0,q,:]

  // stage one 64-key K/V tile into LDS buffer `buf` (4 x 16B per thread)
  auto stage = [&](int buf, int kt2) {
#pragma unroll
    for (int i = 0; i < 2; ++i) {
      int c = i * 256 + tid;            // 0..511 chunks of 16B
      int row = c >> 3, sub = (c & 7) * 8;
      async_ld16((uint32_t)(uintptr_t)&k_tile[buf][row][sub],
                 Kp + (size_t)(kt2 + row) * D_ + sub);
      async_ld16((uint32_t)(uintptr_t)&v_tile[buf][row][sub],
                 Vp + (size_t)row * S_ + kt2 + sub);
    }
  };

  // Q A-fragments for d-chunks [0,32) and [32,64)
  v16bf aq[2];
#pragma unroll
  for (int c = 0; c < 2; ++c) {
    const __bf16* row = Qp + (size_t)ln * D_ + c * 32;
#pragma unroll
    for (int v = 0; v < 8; ++v) {
      bf16x2 p = *(const bf16x2*)(row + a_kofs(g, 2 * v));
      aq[c][2 * v] = p[0]; aq[c][2 * v + 1] = p[1];
    }
  }

  float mi[8], li[8];
  v8f acc[4] = {};
#pragma unroll
  for (int r = 0; r < 8; ++r) { mi[r] = -3.0e38f; li[r] = 0.0f; }

  stage(0, 0);
  int cur = 0;
  for (int kt = 0; kt < S_; kt += 64) {
    wait_async0();          // this thread's copies into buf[cur] landed
    __syncthreads();        // whole block's copies visible; prior reads done
    if (kt + 64 < S_) stage(cur ^ 1, kt + 64);   // overlap next copy w/ compute

    // ---- scores: four 16x16 tiles over the 64-key window (K from LDS) ----
    v8f sc[4];
#pragma unroll
    for (int j = 0; j < 4; ++j) {
      sc[j] = (v8f){};
#pragma unroll
      for (int c = 0; c < 2; ++c) {
        v16bf bk;
        const __bf16* krow = &k_tile[cur][j * 16 + ln][c * 32];
#pragma unroll
        for (int v = 0; v < 8; ++v) {
          bf16x2 p = *(const bf16x2*)(krow + a_kofs(g, 2 * v));
          bk[2 * v] = p[0]; bk[2 * v + 1] = p[1];
        }
        sc[j] = wmma_bf16(aq[c], bk, sc[j]);
      }
    }

    // ---- scale + mask + online softmax (rows live in 16-lane half-waves) ----
#pragma unroll
    for (int r = 0; r < 8; ++r) {
      const float* mrow = Mp + (size_t)(g * 8 + r) * S_ + kt;
      __builtin_prefetch(mrow + 64 + ln, 0, 3);   // next tile's mask, WGP scope
      float s0 = sc[0][r] * 0.125f + mrow[ln];
      float s1 = sc[1][r] * 0.125f + mrow[16 + ln];
      float s2 = sc[2][r] * 0.125f + mrow[32 + ln];
      float s3 = sc[3][r] * 0.125f + mrow[48 + ln];
      float vmx = fmaxf(fmaxf(s0, s1), fmaxf(s2, s3));
      vmx = fmaxf(vmx, __shfl_xor(vmx, 1));
      vmx = fmaxf(vmx, __shfl_xor(vmx, 2));
      vmx = fmaxf(vmx, __shfl_xor(vmx, 4));
      vmx = fmaxf(vmx, __shfl_xor(vmx, 8));
      float mnew = fmaxf(mi[r], vmx);
      float p0 = __expf(s0 - mnew);
      float p1 = __expf(s1 - mnew);
      float p2 = __expf(s2 - mnew);
      float p3 = __expf(s3 - mnew);
      float corr = __expf(mi[r] - mnew);
      float rs = (p0 + p1) + (p2 + p3);
      rs += __shfl_xor(rs, 1);
      rs += __shfl_xor(rs, 2);
      rs += __shfl_xor(rs, 4);
      rs += __shfl_xor(rs, 8);
      li[r] = li[r] * corr + rs;
      mi[r] = mnew;
      acc[0][r] = acc[0][r] * corr;
      acc[1][r] = acc[1][r] * corr;
      acc[2][r] = acc[2][r] * corr;
      acc[3][r] = acc[3][r] * corr;
      __bf16* pr = &p_tile[wave][g * 8 + r][0];
      pr[ln]      = (__bf16)p0;
      pr[16 + ln] = (__bf16)p1;
      pr[32 + ln] = (__bf16)p2;
      pr[48 + ln] = (__bf16)p3;
    }
    __builtin_amdgcn_wave_barrier();   // wave-private P slab; order RAW

    // ---- P (A-frags from LDS) x V (B-frags from LDS, transposed V) ----
#pragma unroll
    for (int kc = 0; kc < 2; ++kc) {
      v16bf ap;
      const __bf16* prow = &p_tile[wave][ln][kc * 32];
#pragma unroll
      for (int v = 0; v < 8; ++v) {
        bf16x2 p = *(const bf16x2*)(prow + a_kofs(g, 2 * v));
        ap[2 * v] = p[0]; ap[2 * v + 1] = p[1];
      }
#pragma unroll
      for (int t = 0; t < 4; ++t) {
        v16bf bv;
        const __bf16* vrow = &v_tile[cur][t * 16 + ln][kc * 32];
#pragma unroll
        for (int v = 0; v < 8; ++v) {
          bf16x2 p = *(const bf16x2*)(vrow + a_kofs(g, 2 * v));
          bv[2 * v] = p[0]; bv[2 * v + 1] = p[1];
        }
        acc[t] = wmma_bf16(ap, bv, acc[t]);
      }
    }

    __syncthreads();        // everyone done reading buf[cur] before reuse
    cur ^= 1;
  }

  // ---- normalize and emit ctx as bf16 [B,S,H] (head-interleaved) ----
#pragma unroll
  for (int t = 0; t < 4; ++t) {
#pragma unroll
    for (int r = 0; r < 8; ++r) {
      int q = tq + g * 8 + r;
      float o = acc[t][r] / li[r];
      ctx[((size_t)b * S_ + q) * H_ + h * D_ + t * 16 + ln] = (__bf16)o;
    }
  }
}

// ---------------------------------------------------------------------------
// Stage 5: residual + layernorm, one 256-thread block per row (H=256)
// ---------------------------------------------------------------------------
__global__ void __launch_bounds__(256)
layernorm_k(const float* __restrict__ hidden, const float* __restrict__ x,
            const float* __restrict__ gamma, const float* __restrict__ beta,
            float* __restrict__ out) {
  __shared__ float red[8];
  const int row = blockIdx.x, t = threadIdx.x;
  const int lane = t & 31, w = t >> 5;
  float v = hidden[(size_t)row * H_ + t] + x[(size_t)row * H_ + t];

  float s = v;
  s += __shfl_xor(s, 1);  s += __shfl_xor(s, 2);  s += __shfl_xor(s, 4);
  s += __shfl_xor(s, 8);  s += __shfl_xor(s, 16);
  if (lane == 0) red[w] = s;
  __syncthreads();
  float mean = 0.0f;
#pragma unroll
  for (int i = 0; i < 8; ++i) mean += red[i];
  mean *= (1.0f / H_);
  __syncthreads();

  float d = v - mean;
  s = d * d;
  s += __shfl_xor(s, 1);  s += __shfl_xor(s, 2);  s += __shfl_xor(s, 4);
  s += __shfl_xor(s, 8);  s += __shfl_xor(s, 16);
  if (lane == 0) red[w] = s;
  __syncthreads();
  float var = 0.0f;
#pragma unroll
  for (int i = 0; i < 8; ++i) var += red[i];
  var *= (1.0f / H_);

  out[(size_t)row * H_ + t] = d * rsqrtf(var + 1e-12f) * gamma[t] + beta[t];
}

// ---------------------------------------------------------------------------
// kernel_launch
// inputs: 0 input, 1 attr, 2 pos, 3 mask, 4 Wq, 5 bq, 6 Wk, 7 bk,
//         8 Wv, 9 bv, 10 Wd, 11 bd, 12 gamma, 13 beta
// ---------------------------------------------------------------------------
extern "C" void kernel_launch(void* const* d_in, const int* in_sizes, int n_in,
                              void* d_out, int out_size, void* d_ws, size_t ws_size,
                              hipStream_t stream) {
  (void)in_sizes; (void)n_in; (void)out_size; (void)ws_size;
  const float* x    = (const float*)d_in[0];
  const float* attr = (const float*)d_in[1];
  const float* pos  = (const float*)d_in[2];
  const float* mask = (const float*)d_in[3];
  const float* Wq = (const float*)d_in[4];  const float* bq = (const float*)d_in[5];
  const float* Wk = (const float*)d_in[6];  const float* bk = (const float*)d_in[7];
  const float* Wv = (const float*)d_in[8];  const float* bv = (const float*)d_in[9];
  const float* Wd = (const float*)d_in[10]; const float* bd = (const float*)d_in[11];
  const float* gamma = (const float*)d_in[12];
  const float* beta  = (const float*)d_in[13];
  float* out = (float*)d_out;

  // workspace carve-up
  char* p = (char*)d_ws;
  __bf16* mixed = (__bf16*)p;             p += XE * 2;
  __bf16* xb    = (__bf16*)p;             p += XE * 2;
  __bf16* Wq_t  = (__bf16*)p;             p += (size_t)H_ * H_ * 2;
  __bf16* Wk_t  = (__bf16*)p;             p += (size_t)H_ * H_ * 2;
  __bf16* Wv_t  = (__bf16*)p;             p += (size_t)H_ * H_ * 2;
  __bf16* Wd_t  = (__bf16*)p;             p += (size_t)H_ * H_ * 2;
  __bf16* Qb    = (__bf16*)p;             p += XE * 2;
  __bf16* Kb    = (__bf16*)p;             p += XE * 2;
  __bf16* Vtb   = (__bf16*)p;             p += XE * 2;
  __bf16* ctxb  = (__bf16*)p;             p += XE * 2;
  float*  hid   = (float*)p;              p += XE * 4;

  // stage 0
  prep_inputs_k<<<(int)((XE + 255) / 256), 256, 0, stream>>>(x, attr, pos, mixed, xb, (int)XE);
  prep_weight_k<<<256, 256, 0, stream>>>(Wq, Wq_t);
  prep_weight_k<<<256, 256, 0, stream>>>(Wk, Wk_t);
  prep_weight_k<<<256, 256, 0, stream>>>(Wv, Wv_t);
  prep_weight_k<<<256, 256, 0, stream>>>(Wd, Wd_t);

  // stage 1: QKV projections (32x16 tile/wave: 512*16 waves -> 1024 blocks)
  const int gemm_blocks = (int)((MROWS / 32) * (H_ / 16) / 8);
  gemm_wmma_k<0><<<gemm_blocks, 256, 0, stream>>>(mixed, Wq_t, bq, (void*)Qb);
  gemm_wmma_k<0><<<gemm_blocks, 256, 0, stream>>>(mixed, Wk_t, bk, (void*)Kb);
  gemm_wmma_k<1><<<gemm_blocks, 256, 0, stream>>>(xb,    Wv_t, bv, (void*)Vtb);

  // stage 2: attention (B*NH * S/128 = 512 blocks)
  flash_attn_k<<<B_ * NH_ * (S_ / 128), 256, 0, stream>>>(Qb, Kb, Vtb, mask, ctxb);

  // stage 4: output projection -> f32 hidden
  gemm_wmma_k<2><<<gemm_blocks, 256, 0, stream>>>(ctxb, Wd_t, bd, (void*)hid);

  // stage 5: residual + layernorm
  layernorm_k<<<(int)MROWS, 256, 0, stream>>>(hid, x, gamma, beta, out);
}
